// MultiHeadSelfAttention_56659208569485
// MI455X (gfx1250) — matrix operand order repaired and checked by reading, not compiled, not run on hardware
//
#include <hip/hip_runtime.h>

// MHSA on gfx1250 (MI455X): bf16 WMMA everywhere, f32 accumulate,
// transposed flash-attention so softmax + P^T fragment build need only
// shfl_xor(16) within the wave32 (no LDS transpose).
// Round 2: raw v_exp_f32 (args <= 0, FTZ underflow is desired) and
// 32x64 GEMM wave tiles (1.5 b128 loads per WMMA instead of 2.5).

#define HIDDEN 1024
#define NHEADS 16
#define HDIM   64
#define SEQ    2048
#define BATCH  2

typedef __attribute__((ext_vector_type(16))) __bf16 v16bf;
typedef __attribute__((ext_vector_type(8)))  __bf16 v8bf;
typedef __attribute__((ext_vector_type(8)))  float  v8f;

union V16U { v16bf v; v8bf h[2]; };

static __device__ __forceinline__ unsigned short f2bf_bits(float f) {
  union { float f; unsigned u; } x; x.f = f;
  unsigned r = (x.u + 0x7FFFu + ((x.u >> 16) & 1u)) >> 16;  // RNE
  return (unsigned short)r;
}

// Raw v_exp_f32: no subnormal-range fixup. Safe here: argument is always
// <= 0 and flush-to-zero on deep underflow is exactly what softmax wants.
static __device__ __forceinline__ float fast_exp2(float x) {
#if __has_builtin(__builtin_amdgcn_exp2f)
  return __builtin_amdgcn_exp2f(x);
#else
  return exp2f(x);
#endif
}

// ---------------- f32 -> bf16 conversion ----------------
__global__ void cvt_f32_bf16(const float* __restrict__ src,
                             unsigned short* __restrict__ dst, int n) {
  int i = blockIdx.x * blockDim.x + threadIdx.x;
  int stride = gridDim.x * blockDim.x;
  for (; i < n; i += stride) dst[i] = f2bf_bits(src[i]);
}

// ---------------- generic GEMM: Y = X @ W^T + b ----------------
// X: [M][1024] bf16, W: [1024][1024] bf16 (row n holds weights for output n)
// Wave tile: 32(M) x 64(N). MODE 0: bf16 row-major out; MODE 1: bf16 V^T
// per-head out; MODE 2: f32 out.
template <int MODE>
__global__ void gemm_bias_kernel(const unsigned short* __restrict__ X,
                                 const unsigned short* __restrict__ W,
                                 const float* __restrict__ bias,
                                 void* __restrict__ out,
                                 int M, float scale) {
  const int wid  = blockIdx.x * (blockDim.x >> 5) + (threadIdx.x >> 5);
  const int lane = threadIdx.x & 31;
  const int m0 = (wid >> 4) * 32;          // 16 N-tiles of 64 across HIDDEN
  const int n0 = (wid & 15) * 64;
  if (m0 >= M) return;
  const int hi   = lane >> 4;              // half-wave select
  const int lq   = lane & 15;
  const int offA = hi ? 16 : 0;

  v8f acc[2][4];
  #pragma unroll
  for (int mi = 0; mi < 2; ++mi)
    #pragma unroll
    for (int t = 0; t < 4; ++t)
      #pragma unroll
      for (int r = 0; r < 8; ++r) acc[mi][t][r] = 0.0f;

  const unsigned short* aRow0 = X + (size_t)(m0 + lq) * HIDDEN;
  const unsigned short* aRow1 = X + (size_t)(m0 + 16 + lq) * HIDDEN;
  for (int k0 = 0; k0 < HIDDEN; k0 += 32) {
    V16U a0, a1;  // A k order used here (contiguous): K = off..off+15 (off = 16*hi)
    a0.h[0] = *reinterpret_cast<const v8bf*>(aRow0 + k0 + offA);
    a0.h[1] = *reinterpret_cast<const v8bf*>(aRow0 + k0 + offA + 8);
    a1.h[0] = *reinterpret_cast<const v8bf*>(aRow1 + k0 + offA);
    a1.h[1] = *reinterpret_cast<const v8bf*>(aRow1 + k0 + offA + 8);
    #pragma unroll
    for (int t = 0; t < 4; ++t) {
      const unsigned short* bRow =
          W + (size_t)(n0 + t * 16 + lq) * HIDDEN + k0 + (hi ? 16 : 0);
      V16U b;  // B k order used here (contiguous, as A): K = j + 16*hi, N = lane&15
      b.h[0] = *reinterpret_cast<const v8bf*>(bRow);
      b.h[1] = *reinterpret_cast<const v8bf*>(bRow + 8);
      acc[0][t] = __builtin_amdgcn_wmma_f32_16x16x32_bf16(
          false, a0.v, false, b.v, (short)0, acc[0][t], false, false);
      acc[1][t] = __builtin_amdgcn_wmma_f32_16x16x32_bf16(
          false, a1.v, false, b.v, (short)0, acc[1][t], false, false);
    }
  }

  #pragma unroll
  for (int mi = 0; mi < 2; ++mi) {
    #pragma unroll
    for (int t = 0; t < 4; ++t) {
      const int n  = n0 + t * 16 + lq;
      const float bs = bias[n];
      #pragma unroll
      for (int r = 0; r < 8; ++r) {
        const int gm = m0 + mi * 16 + r + hi * 8;  // C layout: M = r + 8*hi
        const float v = (acc[mi][t][r] + bs) * scale;
        if (MODE == 0) {
          ((unsigned short*)out)[(size_t)gm * HIDDEN + n] = f2bf_bits(v);
        } else if (MODE == 1) {            // V^T per head: Vt[(bh*64+d)*SEQ + l]
          const int hh = n >> 6, dd = n & 63;
          const int bb = gm >> 11, ll = gm & 2047;
          ((unsigned short*)out)[((size_t)(bb * NHEADS + hh) * HDIM + dd) * SEQ + ll] =
              f2bf_bits(v);
        } else {
          ((float*)out)[(size_t)gm * HIDDEN + n] = v;
        }
      }
    }
  }
}

// ---------------- transposed flash attention ----------------
// One wave handles 16 queries of one (b,h) over all 2048 keys.
// S^T = K @ Q^T (M=keys, N=queries); ctx^T = V^T @ P^T (M=dims, N=queries).
__global__ void attn_kernel(const unsigned short* __restrict__ Q,   // pre-scaled
                            const unsigned short* __restrict__ K,
                            const unsigned short* __restrict__ Vt,  // [bh][d][l]
                            unsigned short* __restrict__ Ctx) {
  const int wid  = blockIdx.x * (blockDim.x >> 5) + (threadIdx.x >> 5);
  if (wid >= BATCH * NHEADS * (SEQ / 16)) return;
  const int lane = threadIdx.x & 31;
  const int bh = wid >> 7;                 // 128 q-tiles per (b,h)
  const int qt = wid & 127;
  const int b = bh >> 4, h = bh & 15;
  const int hi = lane >> 4, lq = lane & 15;
  const int offA = hi ? 16 : 0;

  // Resident Q^T B-fragments (dims 0-31 and 32-63)
  const unsigned short* qrow =
      Q + (size_t)(b * SEQ + qt * 16 + lq) * HIDDEN + h * HDIM;
  V16U bq0, bq1;
  bq0.h[0] = *reinterpret_cast<const v8bf*>(qrow + hi * 16);
  bq0.h[1] = *reinterpret_cast<const v8bf*>(qrow + hi * 16 + 8);
  bq1.h[0] = *reinterpret_cast<const v8bf*>(qrow + 32 + hi * 16);
  bq1.h[1] = *reinterpret_cast<const v8bf*>(qrow + 32 + hi * 16 + 8);

  v8f ctx[4];
  #pragma unroll
  for (int t = 0; t < 4; ++t)
    #pragma unroll
    for (int r = 0; r < 8; ++r) ctx[t][r] = 0.0f;
  float m = -3.0e38f, lsum = 0.0f;

  const unsigned short* kbase = K + (size_t)b * SEQ * HIDDEN + h * HDIM;
  const unsigned short* vbase = Vt + (size_t)bh * HDIM * SEQ;

  for (int kb = 0; kb < SEQ; kb += 32) {
    v8f sc[2];
    #pragma unroll
    for (int t = 0; t < 2; ++t)
      #pragma unroll
      for (int r = 0; r < 8; ++r) sc[t][r] = 0.0f;

    #pragma unroll
    for (int t = 0; t < 2; ++t) {          // key tiles kb..+15, kb+16..+31
      const unsigned short* krow = kbase + (size_t)(kb + t * 16 + lq) * HIDDEN;
      V16U a0, a1;
      a0.h[0] = *reinterpret_cast<const v8bf*>(krow + offA);
      a0.h[1] = *reinterpret_cast<const v8bf*>(krow + offA + 8);
      a1.h[0] = *reinterpret_cast<const v8bf*>(krow + 32 + offA);
      a1.h[1] = *reinterpret_cast<const v8bf*>(krow + 32 + offA + 8);
      sc[t] = __builtin_amdgcn_wmma_f32_16x16x32_bf16(
          false, a0.v, false, bq0.v, (short)0, sc[t], false, false);
      sc[t] = __builtin_amdgcn_wmma_f32_16x16x32_bf16(
          false, a1.v, false, bq1.v, (short)0, sc[t], false, false);
    }

    // Online softmax in log2 domain over the 32-key chunk (per query = lane&15)
    float tmax = sc[0][0];
    #pragma unroll
    for (int t = 0; t < 2; ++t)
      #pragma unroll
      for (int r = 0; r < 8; ++r) tmax = fmaxf(tmax, sc[t][r]);
    tmax = fmaxf(tmax, __shfl_xor(tmax, 16, 32));
    const float mnew  = fmaxf(m, tmax);
    const float alpha = fast_exp2(m - mnew);
    float psum = 0.0f;
    #pragma unroll
    for (int t = 0; t < 2; ++t)
      #pragma unroll
      for (int r = 0; r < 8; ++r) {
        const float p = fast_exp2(sc[t][r] - mnew);
        sc[t][r] = p;
        psum += p;
      }
    psum += __shfl_xor(psum, 16, 32);
    lsum = lsum * alpha + psum;
    m = mnew;

    // Build P^T B-fragment: lane<16 -> keys 0..15, lane>=16 -> keys 16..31
    V16U pb;
    #pragma unroll
    for (int r = 0; r < 8; ++r) {
      const float t0  = sc[0][r], t1 = sc[1][r];
      const float t0x = __shfl_xor(t0, 16, 32);
      const float t1x = __shfl_xor(t1, 16, 32);
      pb.v[r]     = (__bf16)(hi ? t1x : t0);
      pb.v[8 + r] = (__bf16)(hi ? t1 : t0x);
    }

    // ctx^T = alpha*ctx^T + V^T @ P^T  (4 dim-tiles of 16)
    #pragma unroll
    for (int dt = 0; dt < 4; ++dt) {
      const unsigned short* vrow =
          vbase + (size_t)(dt * 16 + lq) * SEQ + kb + offA;
      V16U av;
      av.h[0] = *reinterpret_cast<const v8bf*>(vrow);
      av.h[1] = *reinterpret_cast<const v8bf*>(vrow + 8);
      ctx[dt] = ctx[dt] * alpha;
      ctx[dt] = __builtin_amdgcn_wmma_f32_16x16x32_bf16(
          false, av.v, false, pb.v, (short)0, ctx[dt], false, false);
    }
  }

  const float inv = 1.0f / lsum;
  const size_t orow = (size_t)(b * SEQ + qt * 16 + lq) * HIDDEN + h * HDIM;
  #pragma unroll
  for (int dt = 0; dt < 4; ++dt)
    #pragma unroll
    for (int r = 0; r < 8; ++r)
      Ctx[orow + dt * 16 + r + hi * 8] = f2bf_bits(ctx[dt][r] * inv);
}

// ---------------- launch ----------------
extern "C" void kernel_launch(void* const* d_in, const int* in_sizes, int n_in,
                              void* d_out, int out_size, void* d_ws, size_t ws_size,
                              hipStream_t stream) {
  const float* hs = (const float*)d_in[0];
  const float* Wq = (const float*)d_in[1]; const float* bq = (const float*)d_in[2];
  const float* Wk = (const float*)d_in[3]; const float* bk = (const float*)d_in[4];
  const float* Wv = (const float*)d_in[5]; const float* bv = (const float*)d_in[6];
  const float* Wo = (const float*)d_in[7]; const float* bo = (const float*)d_in[8];

  char* ws = (char*)d_ws;
  const size_t MB = 1u << 20;
  unsigned short* Xbf  = (unsigned short*)(ws + 0 * MB);   // 8 MB
  unsigned short* Wqb  = (unsigned short*)(ws + 8 * MB);   // 2 MB
  unsigned short* Wkb  = (unsigned short*)(ws + 10 * MB);  // 2 MB
  unsigned short* Wvb  = (unsigned short*)(ws + 12 * MB);  // 2 MB
  unsigned short* Wob  = (unsigned short*)(ws + 14 * MB);  // 2 MB
  unsigned short* Qbf  = (unsigned short*)(ws + 16 * MB);  // 8 MB
  unsigned short* Kbf  = (unsigned short*)(ws + 24 * MB);  // 8 MB
  unsigned short* Vt   = (unsigned short*)(ws + 32 * MB);  // 8 MB
  unsigned short* Cbf  = (unsigned short*)(ws + 40 * MB);  // 8 MB

  const int Mrows = BATCH * SEQ;                  // 4096
  const int nX = Mrows * HIDDEN, nW = HIDDEN * HIDDEN;
  cvt_f32_bf16<<<1024, 256, 0, stream>>>(hs, Xbf, nX);
  cvt_f32_bf16<<<512, 256, 0, stream>>>(Wq, Wqb, nW);
  cvt_f32_bf16<<<512, 256, 0, stream>>>(Wk, Wkb, nW);
  cvt_f32_bf16<<<512, 256, 0, stream>>>(Wv, Wvb, nW);
  cvt_f32_bf16<<<512, 256, 0, stream>>>(Wo, Wob, nW);

  // (M/32)*(1024/64) = 2048 wave-tiles, 8 waves/block -> 256 blocks
  const float qscale = 0.125f * 1.4426950408889634f;  // 1/sqrt(64) * log2(e)
  gemm_bias_kernel<0><<<256, 256, 0, stream>>>(Xbf, Wqb, bq, Qbf, Mrows, qscale);
  gemm_bias_kernel<0><<<256, 256, 0, stream>>>(Xbf, Wkb, bk, Kbf, Mrows, 1.0f);
  gemm_bias_kernel<1><<<256, 256, 0, stream>>>(Xbf, Wvb, bv, Vt,  Mrows, 1.0f);

  // 32 (b,h) * 128 q-tiles = 4096 waves -> 512 blocks of 8 waves
  attn_kernel<<<512, 256, 0, stream>>>(Qbf, Kbf, Vt, Cbf);

  gemm_bias_kernel<2><<<256, 256, 0, stream>>>(Cbf, Wob, bo, d_out, Mrows, 1.0f);
}
